// SparseMoEBlock_9328668967107
// MI455X (gfx1250) — matrix operand order repaired and checked by reading, not compiled, not run on hardware
//
#include <hip/hip_runtime.h>
#include <hip/hip_bf16.h>
#include <math.h>

// ---------------- problem constants (from reference setup_inputs) -----------
#define BD   4
#define SD   1024
#define DD   1024          // model dim
#define ED   8             // experts
#define HD   4096          // hidden dim
#define ST   (BD*SD)       // total tokens S = 4096
#define NBIN 16384         // histogram bins over score in [0,1]
#define PAIRS_MAX 9216     // k (=8192) + slack for threshold-bin overshoot

typedef __attribute__((ext_vector_type(16))) _Float16 v16h;
typedef __attribute__((ext_vector_type(8)))  _Float16 v8h;
typedef __attribute__((ext_vector_type(2)))  __fp16   v2fp;   // cvt_pkrtz result type
typedef __attribute__((ext_vector_type(8)))  float    v8f;

// ---------------- workspace layout (bytes) ----------------------------------
#define WS_SCORES  0
#define WS_HIST    (WS_SCORES + ED*ST*4)
#define WS_META    (WS_HIST + NBIN*4)
#define WS_PTOK    (WS_META + 256)
#define WS_PSCL    (WS_PTOK + PAIRS_MAX*4)
#define WS_ABUF    ((WS_PSCL + PAIRS_MAX*4 + 255) & ~255)
#define WS_HBUF    (WS_ABUF + PAIRS_MAX*DD*2)

__device__ __forceinline__ float gelu_tanh(float x) {
    float x3 = x * x * x;
    return 0.5f * x * (1.0f + tanhf(0.7978845608028654f * (x + 0.044715f * x3)));
}

// ---------------- 1) router: logits + softmax over experts ------------------
__global__ void moe_router(const float* __restrict__ x,
                           const float* __restrict__ gw,
                           const float* __restrict__ eb,
                           float* __restrict__ scores) {
    const int s = blockIdx.x;
    const int tid = threadIdx.x;               // 256 threads
    float part[ED];
#pragma unroll
    for (int e = 0; e < ED; ++e) part[e] = 0.f;
    for (int d = tid; d < DD; d += 256) {
        float xv = x[(size_t)s * DD + d];
#pragma unroll
        for (int e = 0; e < ED; ++e) part[e] += xv * gw[(size_t)e * DD + d];
    }
    __shared__ float red[256];
    __shared__ float logit[ED];
    for (int e = 0; e < ED; ++e) {
        red[tid] = part[e];
        __syncthreads();
        for (int off = 128; off > 0; off >>= 1) {
            if (tid < off) red[tid] += red[tid + off];
            __syncthreads();
        }
        if (tid == 0) logit[e] = red[0] + eb[e];
        __syncthreads();
    }
    if (tid == 0) {
        float m = logit[0];
#pragma unroll
        for (int e = 1; e < ED; ++e) m = fmaxf(m, logit[e]);
        float sum = 0.f, p[ED];
#pragma unroll
        for (int e = 0; e < ED; ++e) { p[e] = __expf(logit[e] - m); sum += p[e]; }
        float inv = 1.f / sum;
#pragma unroll
        for (int e = 0; e < ED; ++e) scores[(size_t)e * ST + s] = p[e] * inv;
    }
}

// ---------------- 2) histogram of scores ------------------------------------
__global__ void moe_hist(const float* __restrict__ scores, unsigned* __restrict__ hist) {
    int i = blockIdx.x * blockDim.x + threadIdx.x;
    float v = scores[i];
    int b = (int)(v * (float)NBIN);
    b = min(max(b, 0), NBIN - 1);
    atomicAdd(&hist[b], 1u);
}

// ---------------- 3) pick threshold (top-k), zero counters ------------------
__global__ void moe_thresh(const unsigned* __restrict__ hist,
                           const int* __restrict__ capacity,
                           int* __restrict__ meta) {
    const int K = ST * capacity[0];
    unsigned acc = 0;
    int b = 0;
    for (int i = NBIN - 1; i >= 0; --i) {
        acc += hist[i];
        if ((int)acc >= K) { b = i; break; }
    }
    ((float*)meta)[0] = (float)b / (float)NBIN;
    for (int e = 0; e < ED; ++e) { meta[8 + e] = 0; meta[32 + e] = 0; }
}

// ---------------- 4) per-expert counts --------------------------------------
__global__ void moe_count(const float* __restrict__ scores, int* __restrict__ meta) {
    int i = blockIdx.x * blockDim.x + threadIdx.x;
    float th = ((const float*)meta)[0];
    if (scores[i] >= th) atomicAdd(&meta[8 + i / ST], 1);
}

// ---------------- 5) exclusive scan -> segment offsets ----------------------
__global__ void moe_offsets(int* __restrict__ meta) {
    int run = 0;
    for (int e = 0; e < ED; ++e) {
        meta[16 + e] = run;
        int c = meta[8 + e];
        if (run + c > PAIRS_MAX) c = PAIRS_MAX - run;
        run += c;
        meta[32 + e] = 0;
    }
    meta[16 + ED] = run;
}

// ---------------- 6) append selected pairs ----------------------------------
__global__ void moe_append(const float* __restrict__ scores, int* __restrict__ meta,
                           int* __restrict__ ptok, float* __restrict__ pscl) {
    int i = blockIdx.x * blockDim.x + threadIdx.x;
    float th = ((const float*)meta)[0];
    float v = scores[i];
    if (v >= th) {
        int e = i / ST, s = i % ST;
        int slot = meta[16 + e] + atomicAdd(&meta[32 + e], 1);
        if (slot < meta[16 + e + 1]) { ptok[slot] = s; pscl[slot] = v; }
    }
}

// ---------------- 7) gather routed token rows -> f16 A buffer ---------------
__global__ void moe_gather(const float* __restrict__ x, const int* __restrict__ meta,
                           const int* __restrict__ ptok, _Float16* __restrict__ abuf) {
    int p = blockIdx.x;
    if (p >= meta[16 + ED]) return;
    int tok = ptok[p];
    for (int d = threadIdx.x * 2; d < DD; d += 512) {
        v2fp h = __builtin_amdgcn_cvt_pkrtz(x[(size_t)tok * DD + d],
                                            x[(size_t)tok * DD + d + 1]);
        *(v2fp*)&abuf[(size_t)p * DD + d] = h;
    }
}

// ---------------- 8) fused tiled WMMA GEMM (both layers) --------------------
// mode 0:  Hbuf = gelu(A(Abuf) @ w1[e] + b1[e]) * combine        (f16 out)
// mode 1:  out[tok] += A(Hbuf) @ w2[e] + combine * b2[e]         (atomic f32)
// Block tile: 128(M) x 128(N), K-step 32, double-buffered LDS,
// 8 waves (2x4), each wave 64x32 -> 8 wmma per K-step.
__global__ __launch_bounds__(256)
void moe_gemm(const _Float16* __restrict__ A, int lda, int Kdim,
              const float* __restrict__ Bw, int Ndim,
              const float* __restrict__ bias,
              const int* __restrict__ meta,
              const int* __restrict__ ptok, const float* __restrict__ pscl,
              _Float16* __restrict__ hbuf, float* __restrict__ out, int mode) {
    const int e = blockIdx.z;
    const int rbase = meta[16 + e];
    const int cnt = meta[16 + e + 1] - rbase;
    const int mb = blockIdx.y * 128;
    if (mb >= cnt) return;
    const int nb = blockIdx.x * 128;

    const float* Be = Bw + (size_t)e * Kdim * Ndim;
    const float* be = bias + (size_t)e * Ndim;

    // double-buffered tiles: A as [m][k] in natural K order, B as [n][k]
    __shared__ alignas(64) _Float16 lA[2][128][32];   // 2 x 8 KB
    __shared__ alignas(64) _Float16 lB[2][128][32];   // 2 x 8 KB

    const int tid  = threadIdx.x;
    const int lane = tid & 31;
    const int wave = tid >> 5;
    const int wm = wave >> 2, wn = wave & 3;          // 2x4 wave grid
    const int half = lane >> 4, lr = lane & 15;

    // --- staging thread mappings (fixed for the whole kernel)
    const int arow = tid >> 1;                        // A: 128 rows, 2 thr/row
    const int aseg0 = (tid & 1) * 2;                  // segs {0,1} or {2,3}
    const int bk = (tid >> 4) * 2;                    // B: k-pair base (0..30)
    const int bn0 = (tid & 15) * 8;                   // 8 columns per thread

    v8f acc[4][2];
#pragma unroll
    for (int i = 0; i < 4; ++i)
#pragma unroll
        for (int j = 0; j < 2; ++j) acc[i][j] = (v8f)0.f;

    v8h ra[2];          // staged A regs (2 x 8 f16)
    float4 rb[4];       // staged B regs (2 rows x 8 floats)

    auto loadG = [&](int k0) {
#pragma unroll
        for (int s = 0; s < 2; ++s) {
            int seg = aseg0 + s;
            v8h val = {};
            if (mb + arow < cnt)
                val = *(const v8h*)(A + (size_t)(rbase + mb + arow) * lda + k0 + seg * 8);
            ra[s] = val;
        }
        const float* src0 = Be + (size_t)(k0 + bk) * Ndim + nb + bn0;
        const float* src1 = src0 + Ndim;
        rb[0] = *(const float4*)(src0);
        rb[1] = *(const float4*)(src0 + 4);
        rb[2] = *(const float4*)(src1);
        rb[3] = *(const float4*)(src1 + 4);
        if (k0 + 32 < Kdim)
            __builtin_prefetch(src0 + (size_t)32 * Ndim, 0, 1);   // next K slab
    };

    auto storeL = [&](int buf) {
#pragma unroll
        for (int s = 0; s < 2; ++s) {
            int seg = aseg0 + s;
            int pos = seg * 8;                                    // 0,8,16,24
            *(v8h*)&lA[buf][arow][pos] = ra[s];
        }
        // packed K-pair stores: one b32 per column
        const float* f0 = (const float*)&rb[0];                   // row k
        const float* f1 = (const float*)&rb[2];                   // row k+1
#pragma unroll
        for (int j = 0; j < 8; ++j) {
            v2fp h = __builtin_amdgcn_cvt_pkrtz(f0[j], f1[j]);
            *(v2fp*)&lB[buf][bn0 + j][bk] = h;
        }
    };

    const int KT = Kdim >> 5;
    loadG(0);
    storeL(0);

    for (int kt = 0; kt < KT; ++kt) {
        __syncthreads();
        const int cur = kt & 1;
        const bool more = (kt + 1 < KT);
        if (more) loadG((kt + 1) << 5);               // issue next-tile loads early

        v16h a[4], b[2];
#pragma unroll
        for (int mi = 0; mi < 4; ++mi)
            a[mi] = *(const v16h*)&lA[cur][wm * 64 + mi * 16 + lr][half * 16];
#pragma unroll
        for (int ni = 0; ni < 2; ++ni)
            b[ni] = *(const v16h*)&lB[cur][wn * 32 + ni * 16 + lr][half * 16];
#pragma unroll
        for (int mi = 0; mi < 4; ++mi)
#pragma unroll
            for (int ni = 0; ni < 2; ++ni)
                acc[mi][ni] = __builtin_amdgcn_wmma_f32_16x16x32_f16(
                    false, a[mi], false, b[ni], (short)0, acc[mi][ni], false, false);

        if (more) storeL(cur ^ 1);                    // fill other buffer
    }

    // --- epilogue: C element (vgpr r, lane) = C[r + 8*half][lr] of 16x16 tile
#pragma unroll
    for (int mi = 0; mi < 4; ++mi)
#pragma unroll
        for (int ni = 0; ni < 2; ++ni) {
            int n = nb + wn * 32 + ni * 16 + lr;
#pragma unroll
            for (int r = 0; r < 8; ++r) {
                int gr = mb + wm * 64 + mi * 16 + r + 8 * half;   // row in expert
                if (gr >= cnt) continue;
                float v = acc[mi][ni][r];
                int p = rbase + gr;
                if (mode == 0) {
                    v = gelu_tanh(v + be[n]) * pscl[p];
                    hbuf[(size_t)p * HD + n] = (_Float16)v;
                } else {
                    v += pscl[p] * be[n];
                    atomicAdd(&out[(size_t)ptok[p] * DD + n], v);
                }
            }
        }
}

// ---------------- launch -----------------------------------------------------
extern "C" void kernel_launch(void* const* d_in, const int* in_sizes, int n_in,
                              void* d_out, int out_size, void* d_ws, size_t ws_size,
                              hipStream_t stream) {
    const float* x  = (const float*)d_in[0];
    const float* gw = (const float*)d_in[1];
    const float* eb = (const float*)d_in[2];
    const float* w1 = (const float*)d_in[3];
    const float* b1 = (const float*)d_in[4];
    const float* w2 = (const float*)d_in[5];
    const float* b2 = (const float*)d_in[6];
    const int* cap  = (const int*)d_in[7];
    float* out = (float*)d_out;

    char* ws = (char*)d_ws;
    float*    scores = (float*)(ws + WS_SCORES);
    unsigned* hist   = (unsigned*)(ws + WS_HIST);
    int*      meta   = (int*)(ws + WS_META);
    int*      ptok   = (int*)(ws + WS_PTOK);
    float*    pscl   = (float*)(ws + WS_PSCL);
    _Float16* abuf   = (_Float16*)(ws + WS_ABUF);
    _Float16* hbuf   = (_Float16*)(ws + WS_HBUF);

    (void)hipMemsetAsync(hist, 0, NBIN * sizeof(unsigned), stream);
    (void)hipMemsetAsync(out, 0, (size_t)out_size * sizeof(float), stream);

    moe_router <<<ST, 256, 0, stream>>>(x, gw, eb, scores);
    moe_hist   <<<(ED * ST) / 256, 256, 0, stream>>>(scores, hist);
    moe_thresh <<<1, 1, 0, stream>>>(hist, cap, meta);
    moe_count  <<<(ED * ST) / 256, 256, 0, stream>>>(scores, meta);
    moe_offsets<<<1, 1, 0, stream>>>(meta);
    moe_append <<<(ED * ST) / 256, 256, 0, stream>>>(scores, meta, ptok, pscl);
    moe_gather <<<PAIRS_MAX, 256, 0, stream>>>(x, meta, ptok, abuf);

    // layer 1: [cnt_e,1024] @ [1024,4096] -> gelu*combine -> Hbuf (f16)
    moe_gemm<<<dim3(HD / 128, 32, ED), 256, 0, stream>>>(
        abuf, DD, DD, w1, HD, b1, meta, ptok, pscl, hbuf, out, 0);
    // layer 2: [cnt_e,4096] @ [4096,1024] -> +combine*b2 -> atomic scatter
    moe_gemm<<<dim3(DD / 128, 32, ED), 256, 0, stream>>>(
        hbuf, HD, HD, w2, DD, b2, meta, ptok, pscl, hbuf, out, 1);
}
